// Sum_Attention_50440095924278
// MI455X (gfx1250) — matrix operand order repaired and checked by reading, not compiled, not run on hardware
//
#include <hip/hip_runtime.h>
#include <hip/hip_bf16.h>

// ---------------- types for WMMA ----------------
typedef __bf16 bf16_t;
typedef __attribute__((ext_vector_type(16))) __bf16 v16bf;
typedef __attribute__((ext_vector_type(8)))  float  v8f;

union ABPack { v16bf v; uint4 q[2]; };

static __device__ inline v8f wmma_bf16(v16bf a, v16bf b, v8f c) {
  // D = A(16x32) * B(32x16) + C, f32 accumulate
  return __builtin_amdgcn_wmma_f32_16x16x32_bf16(false, a, false, b, (short)0, c,
                                                 false, false);
}

// ---------------- problem constants ----------------
#define BATCH 8
#define T_SEQ 243
#define T_PAD 256
#define NJ    17
#define CIN   512
#define NH    8
#define HD    64
#define M_TOT (BATCH * T_SEQ * NJ)   // 33048
#define M_PAD 33056                  // next multiple of 32
#define MT2   (M_PAD / 32)           // 1033 (32-row super-tiles)
#define NBHJ  (BATCH * NH * NJ)      // 1088
#define VT_PITCH 264                 // LDS row pitch (bf16 elems), bank-friendly

// ---------------- stage 0: precision / layout prep ----------------
__global__ __launch_bounds__(256) void convert_x_kernel(
    const float* __restrict__ x, bf16_t* __restrict__ xb, int total, int valid) {
  for (int i = blockIdx.x * blockDim.x + threadIdx.x; i < total;
       i += gridDim.x * blockDim.x)
    xb[i] = (i < valid) ? (bf16_t)x[i] : (bf16_t)0.0f;
}

// wt[n*K + k] = w[k*N + n]   (w is [K][N] row-major)
__global__ __launch_bounds__(256) void transpose_w_kernel(
    const float* __restrict__ w, bf16_t* __restrict__ wt, int K, int N) {
  int i = blockIdx.x * blockDim.x + threadIdx.x;
  if (i < N * K) {
    int n = i / K, k = i - n * K;
    wt[i] = (bf16_t)w[k * N + n];
  }
}

// ---------------- stage 1: QKV GEMM (bf16 WMMA), 32x64 per wave -------------
// scatter-stores q/k/v into per-(b,h,j) row-major [256][64] bf16 tiles
__global__ __launch_bounds__(256) void qkv_gemm_kernel(
    const bf16_t* __restrict__ xb, const bf16_t* __restrict__ wt,
    bf16_t* __restrict__ qg, bf16_t* __restrict__ kg, bf16_t* __restrict__ vg) {
  const int tid = threadIdx.x, lane = tid & 31, wave = tid >> 5;
  const int job = blockIdx.x * 8 + wave;       // MT2*24 jobs
  const int mt = job / 24, ng = job - mt * 24;
  const int m0 = mt * 32, nb = ng * 64;
  const int half = lane >> 4, l15 = lane & 15;

  v8f acc[2][4] = {};
  const bf16_t* arow0 = xb + (size_t)(m0 + l15) * CIN;
  const bf16_t* arow1 = arow0 + (size_t)16 * CIN;
#pragma unroll 4
  for (int c = 0; c < 16; ++c) {               // K = 512 = 16 * 32
    ABPack A0, A1;
    const bf16_t* pa0 = arow0 + c * 32 + half * 8;
    const bf16_t* pa1 = arow1 + c * 32 + half * 8;
    A0.q[0] = *(const uint4*)(pa0);
    A0.q[1] = *(const uint4*)(pa0 + 16);
    A1.q[0] = *(const uint4*)(pa1);
    A1.q[1] = *(const uint4*)(pa1 + 16);
#pragma unroll
    for (int jn = 0; jn < 4; ++jn) {
      const bf16_t* pb = wt + (size_t)(nb + jn * 16 + l15) * CIN + c * 32 + half * 8;
      ABPack B;
      B.q[0] = *(const uint4*)(pb);
      B.q[1] = *(const uint4*)(pb + 16);
      acc[0][jn] = wmma_bf16(A0.v, B.v, acc[0][jn]);
      acc[1][jn] = wmma_bf16(A1.v, B.v, acc[1][jn]);
    }
  }
  // scatter store: n = nb + jn*16 + l15 ; s = n>>9, h = (n>>6)&7, d = n&63
  const int s = nb >> 9;
  const int h = (nb >> 6) & 7;
  bf16_t* dst = (s == 0) ? qg : (s == 1) ? kg : vg;
#pragma unroll
  for (int mi = 0; mi < 2; ++mi) {
#pragma unroll
    for (int r = 0; r < 8; ++r) {
      int m = m0 + mi * 16 + r + half * 8;
      if (m < M_TOT) {
        int b = m / (T_SEQ * NJ);
        int rm = m - b * (T_SEQ * NJ);
        int t = rm / NJ, jj = rm - t * NJ;
        int bhj = (b * NH + h) * NJ + jj;
        bf16_t* p = dst + (size_t)bhj * (T_PAD * HD) + t * HD;
#pragma unroll
        for (int jn = 0; jn < 4; ++jn) p[jn * 16 + l15] = (bf16_t)acc[mi][jn][r];
      }
    }
  }
}

// ---------------- stage 2: per-(b,h,j) attention ----------------
// grid = 1088 WGs of 256 threads; wave w handles t-tiles 2w, 2w+1 of 16 rows.
__global__ __launch_bounds__(256) void attn_kernel(
    const bf16_t* __restrict__ qg, const bf16_t* __restrict__ kg,
    const bf16_t* __restrict__ vg, const float* __restrict__ amap,
    const float* __restrict__ wptr, bf16_t* __restrict__ obf) {
  extern __shared__ char smem_raw[];
  bf16_t* vT = (bf16_t*)smem_raw;                                  // [64][VT_PITCH]
  bf16_t* attnBase = (bf16_t*)(smem_raw + (size_t)HD * VT_PITCH * 2);

  const int tid = threadIdx.x, lane = tid & 31, wave = tid >> 5;
  const int half = lane >> 4, l15 = lane & 15;
  const int bhj = blockIdx.x;
  const float wgt = wptr[0];
  const float omw = 1.0f - wgt;

  const bf16_t* q = qg + (size_t)bhj * (T_PAD * HD);
  const bf16_t* k = kg + (size_t)bhj * (T_PAD * HD);
  const bf16_t* v = vg + (size_t)bhj * (T_PAD * HD);
  const float* am = amap + (size_t)bhj * (T_SEQ * T_SEQ);

  // cooperative v -> LDS transpose: vT[d][t]
  for (int i = tid; i < T_PAD * HD; i += 256) {
    int t = i >> 6, d = i & 63;
    vT[d * VT_PITCH + t] = v[i];
  }
  __syncthreads();

  bf16_t* attn = attnBase + (size_t)wave * (16 * VT_PITCH);

  int b = bhj / (NH * NJ);
  int rem = bhj - b * (NH * NJ);
  int h = rem / NJ;
  int jj = rem - h * NJ;
  bf16_t* obase = obf + ((size_t)(b * T_SEQ) * NJ + jj) * CIN + h * HD;

  for (int tt = 0; tt < 2; ++tt) {
    const int t0 = (wave * 2 + tt) * 16;
    // ---- S = (q k^T) : 16 C-tiles covering 16 x 256 ----
    v8f acc[16] = {};
    const bf16_t* qrow = q + (size_t)(t0 + l15) * HD;
#pragma unroll
    for (int c = 0; c < 2; ++c) {              // K = hd = 64 = 2*32
      ABPack A;
      const bf16_t* pa = qrow + c * 32 + half * 8;
      A.q[0] = *(const uint4*)(pa);
      A.q[1] = *(const uint4*)(pa + 16);
#pragma unroll
      for (int j = 0; j < 16; ++j) {
        const bf16_t* pb = k + (size_t)(j * 16 + l15) * HD + c * 32 + half * 8;
        ABPack B;
        B.q[0] = *(const uint4*)(pb);
        B.q[1] = *(const uint4*)(pb + 16);
        acc[j] = wmma_bf16(A.v, B.v, acc[j]);
      }
    }
    // ---- masked softmax over s<243, then blend with att_map ----
    const float scale = 0.125f;                // hd^-0.5
    float rm[8], rs[8];
#pragma unroll
    for (int r = 0; r < 8; ++r) rm[r] = -1e30f;
#pragma unroll
    for (int j = 0; j < 16; ++j) {
      int s = j * 16 + l15;
      if (s < T_SEQ) {
#pragma unroll
        for (int r = 0; r < 8; ++r) rm[r] = fmaxf(rm[r], acc[j][r] * scale);
      }
    }
#pragma unroll
    for (int msk = 1; msk < 16; msk <<= 1) {
#pragma unroll
      for (int r = 0; r < 8; ++r) rm[r] = fmaxf(rm[r], __shfl_xor(rm[r], msk, 32));
    }
#pragma unroll
    for (int r = 0; r < 8; ++r) rs[r] = 0.0f;
#pragma unroll
    for (int j = 0; j < 16; ++j) {
      int s = j * 16 + l15;
#pragma unroll
      for (int r = 0; r < 8; ++r) {
        float p = (s < T_SEQ) ? __expf(acc[j][r] * scale - rm[r]) : 0.0f;
        acc[j][r] = p;
        rs[r] += p;
      }
    }
#pragma unroll
    for (int msk = 1; msk < 16; msk <<= 1) {
#pragma unroll
      for (int r = 0; r < 8; ++r) rs[r] += __shfl_xor(rs[r], msk, 32);
    }
    // blend + write bf16 attn tile to per-wave LDS [16][VT_PITCH]
#pragma unroll
    for (int r = 0; r < 8; ++r) {
      int mrow = r + half * 8;
      int t = t0 + mrow;
      float inv = 1.0f / rs[r];
#pragma unroll
      for (int j = 0; j < 16; ++j) {
        int s = j * 16 + l15;
        float a = 0.0f;
        if (t < T_SEQ && s < T_SEQ)
          a = wgt * (acc[j][r] * inv) + omw * am[t * T_SEQ + s];
        attn[mrow * VT_PITCH + s] = (bf16_t)a;
      }
    }
    __syncthreads();   // make per-wave LDS writes visible cross-lane
    // ---- O = attn @ v : 16 x 64 ----
    v8f oc[4] = {};
#pragma unroll
    for (int c = 0; c < 8; ++c) {              // K = 256 = 8*32
      ABPack A;
      const bf16_t* pa = attn + l15 * VT_PITCH + c * 32 + half * 8;
      A.q[0] = *(const uint4*)(pa);
      A.q[1] = *(const uint4*)(pa + 16);
#pragma unroll
      for (int nd = 0; nd < 4; ++nd) {
        const bf16_t* pb = vT + (size_t)(nd * 16 + l15) * VT_PITCH + c * 32 + half * 8;
        ABPack B;
        B.q[0] = *(const uint4*)(pb);
        B.q[1] = *(const uint4*)(pb + 16);
        oc[nd] = wmma_bf16(A.v, B.v, oc[nd]);
      }
    }
    // store O (bf16) into [b][t][jj][h*64+d] row-major (M x 512)
#pragma unroll
    for (int r = 0; r < 8; ++r) {
      int t = t0 + r + half * 8;
      if (t < T_SEQ) {
        bf16_t* po = obase + (size_t)t * (NJ * CIN);
#pragma unroll
        for (int nd = 0; nd < 4; ++nd) po[nd * 16 + l15] = (bf16_t)oc[nd][r];
      }
    }
    __syncthreads();
  }
}

// ---------------- stage 3: output projection + bias (f32 out), 32x64/wave ---
__global__ __launch_bounds__(256) void proj_gemm_kernel(
    const bf16_t* __restrict__ ob, const bf16_t* __restrict__ wt,
    const float* __restrict__ bias, float* __restrict__ out) {
  const int tid = threadIdx.x, lane = tid & 31, wave = tid >> 5;
  const int job = blockIdx.x * 8 + wave;       // MT2*8 jobs
  const int mt = job >> 3, ng = job & 7;
  const int m0 = mt * 32, nb = ng * 64;
  const int half = lane >> 4, l15 = lane & 15;

  v8f acc[2][4] = {};
  const bf16_t* arow0 = ob + (size_t)(m0 + l15) * CIN;
  const bf16_t* arow1 = arow0 + (size_t)16 * CIN;
#pragma unroll 4
  for (int c = 0; c < 16; ++c) {
    ABPack A0, A1;
    const bf16_t* pa0 = arow0 + c * 32 + half * 8;
    const bf16_t* pa1 = arow1 + c * 32 + half * 8;
    A0.q[0] = *(const uint4*)(pa0);
    A0.q[1] = *(const uint4*)(pa0 + 16);
    A1.q[0] = *(const uint4*)(pa1);
    A1.q[1] = *(const uint4*)(pa1 + 16);
#pragma unroll
    for (int jn = 0; jn < 4; ++jn) {
      const bf16_t* pb = wt + (size_t)(nb + jn * 16 + l15) * CIN + c * 32 + half * 8;
      ABPack B;
      B.q[0] = *(const uint4*)(pb);
      B.q[1] = *(const uint4*)(pb + 16);
      acc[0][jn] = wmma_bf16(A0.v, B.v, acc[0][jn]);
      acc[1][jn] = wmma_bf16(A1.v, B.v, acc[1][jn]);
    }
  }
#pragma unroll
  for (int mi = 0; mi < 2; ++mi) {
#pragma unroll
    for (int r = 0; r < 8; ++r) {
      int m = m0 + mi * 16 + r + half * 8;
      if (m < M_TOT) {
        float* po = out + (size_t)m * CIN + nb;
#pragma unroll
        for (int jn = 0; jn < 4; ++jn) {
          int n = jn * 16 + l15;
          po[n] = acc[mi][jn][r] + bias[nb + n];
        }
      }
    }
  }
}

// ---------------- host side ----------------
extern "C" void kernel_launch(void* const* d_in, const int* in_sizes, int n_in,
                              void* d_out, int out_size, void* d_ws, size_t ws_size,
                              hipStream_t stream) {
  const float* x      = (const float*)d_in[0];
  const float* amap   = (const float*)d_in[1];
  const float* weight = (const float*)d_in[2];
  const float* W_qkv  = (const float*)d_in[3];
  const float* W_proj = (const float*)d_in[4];
  const float* b_proj = (const float*)d_in[5];
  float* out = (float*)d_out;

  // workspace carve-up (bytes, 256-aligned)
  char* ws = (char*)d_ws;
  size_t off = 0;
  auto carve = [&](size_t bytes) {
    char* p = ws + off;
    off = (off + bytes + 255) & ~(size_t)255;
    return p;
  };
  bf16_t* xb     = (bf16_t*)carve((size_t)M_PAD * CIN * 2);        // 33.8 MB
  bf16_t* wqkvT  = (bf16_t*)carve((size_t)(3 * CIN) * CIN * 2);    // 1.5 MB
  bf16_t* wprojT = (bf16_t*)carve((size_t)CIN * CIN * 2);          // 0.5 MB
  bf16_t* qg     = (bf16_t*)carve((size_t)NBHJ * T_PAD * HD * 2);  // 35.7 MB
  bf16_t* kg     = (bf16_t*)carve((size_t)NBHJ * T_PAD * HD * 2);
  bf16_t* vg     = (bf16_t*)carve((size_t)NBHJ * T_PAD * HD * 2);
  bf16_t* obf    = (bf16_t*)carve((size_t)M_PAD * CIN * 2);        // 33.8 MB
  (void)ws_size; (void)in_sizes; (void)n_in; (void)out_size;

  // stage 0: precision & layout prep
  convert_x_kernel<<<4096, 256, 0, stream>>>(x, xb, M_PAD * CIN, M_TOT * CIN);
  transpose_w_kernel<<<(3 * CIN * CIN + 255) / 256, 256, 0, stream>>>(
      W_qkv, wqkvT, CIN, 3 * CIN);
  transpose_w_kernel<<<(CIN * CIN + 255) / 256, 256, 0, stream>>>(
      W_proj, wprojT, CIN, CIN);

  // stage 1: QKV GEMM -> q/k/v tiles (MT2*24 wave jobs, 8 waves/block)
  qkv_gemm_kernel<<<(MT2 * 24) / 8, 256, 0, stream>>>(xb, wqkvT, qg, kg, vg);

  // stage 2: attention (dynamic LDS: v^T + 8 per-wave attn tiles)
  const size_t smem2 = (size_t)HD * VT_PITCH * 2 + (size_t)8 * 16 * VT_PITCH * 2;
  attn_kernel<<<NBHJ, 256, smem2, stream>>>(qg, kg, vg, amap, weight, obf);

  // stage 3: projection + bias -> f32 output (MT2*8 wave jobs)
  proj_gemm_kernel<<<MT2, 256, 0, stream>>>(obf, wprojT, b_proj, out);
}
